// MultiHeadAttention_40553081208984
// MI455X (gfx1250) — matrix operand order repaired and checked by reading, not compiled, not run on hardware
//
#include <hip/hip_runtime.h>
#include <hip/hip_bf16.h>

// ---------------------------------------------------------------------------
// MultiHeadAttention on gfx1250 (CDNA5), wave32 + v_wmma_f32_16x16x32_f16.
// B=1, S=4096, D=512, H=8, Hd=64. Flash-attention: scores never hit memory.
// GEMMs: 32x64 tiles/wave (24 B loaded per WMMA). Attention: K/V tiles staged
// in LDS shared by all 8 waves of a block (same head), double-buffered; the
// staging uses GLOBAL_LOAD_ASYNC_TO_LDS_B128 (ASYNCcnt) when the toolchain
// exposes the builtin, with an explicit s_wait_asynccnt before the barrier.
// ---------------------------------------------------------------------------

#define S_LEN 4096
#define D_DIM 512
#define H_NUM 8
#define HD 64
#define E3 1536   // 3*D

typedef _Float16 h8  __attribute__((ext_vector_type(8)));
typedef _Float16 h16 __attribute__((ext_vector_type(16)));
typedef float    f8  __attribute__((ext_vector_type(8)));
typedef int      i4  __attribute__((ext_vector_type(4)));

#if defined(__has_builtin)
#if __has_builtin(__builtin_amdgcn_global_load_async_to_lds_b128)
#define HAVE_ASYNC_LDS 1
#endif
#if __has_builtin(__builtin_amdgcn_s_wait_asynccnt)
#define HAVE_WAIT_ASYNC_BUILTIN 1
#endif
#endif

static __device__ __forceinline__ h16 cat16(h8 lo, h8 hi) {
    return __builtin_shufflevector(lo, hi, 0,1,2,3,4,5,6,7,8,9,10,11,12,13,14,15);
}

static __device__ __forceinline__ f8 wmma_f16(h16 a, h16 b, f8 c) {
    return __builtin_amdgcn_wmma_f32_16x16x32_f16(
        /*neg_a=*/false, a, /*neg_b=*/false, b,
        /*c_mod=*/(short)0, c, /*reuse_a=*/false, /*reuse_b=*/false);
}

// A-fragment (16x32 f16): lane m = lane&15, klo = (lane>>4)*8.
// halves [0..15] = K 2*klo..2*klo+15 (contiguous).
static __device__ __forceinline__ h16 load_a(const _Float16* rowptr, int klo) {
    h8 lo = *(const h8*)(rowptr + 2 * klo);
    h8 hi = *(const h8*)(rowptr + 2 * klo + 8);
    return cat16(lo, hi);
}

// 16-byte global->LDS copy for staging. Async (direct-to-LDS, ASYNCcnt) when
// available; otherwise through VGPRs. The builtin takes (int4 AS1* src,
// int4 AS3* dst, imm offset, imm cpol) per the toolchain's diagnostic.
// AS3 pointers are 32-bit; the low 32 bits of a generic LDS address are the
// LDS offset, so the integer round-trip yields the correct AS3 pointer.
static __device__ __forceinline__ void stage16(const _Float16* g, _Float16* l) {
#if defined(HAVE_ASYNC_LDS)
    typedef __attribute__((address_space(1))) i4 gi4_t;
    typedef __attribute__((address_space(3))) i4 li4_t;
    __builtin_amdgcn_global_load_async_to_lds_b128(
        (gi4_t*)(uintptr_t)(const void*)g,
        (li4_t*)(uintptr_t)(uint32_t)(uintptr_t)(void*)l,
        0, 0);
#else
    *(h8*)l = *(const h8*)g;
#endif
}

// Wave-local completion of async staging; must precede the publishing barrier.
static __device__ __forceinline__ void stage_wait() {
#if defined(HAVE_ASYNC_LDS)
#if defined(HAVE_WAIT_ASYNC_BUILTIN)
    __builtin_amdgcn_s_wait_asynccnt(0);
#else
    asm volatile("s_wait_asynccnt 0x0" ::: "memory");
#endif
#endif
}

// ---------------------------------------------------------------------------
// Kernel 1: f32 -> f16 conversion
// ---------------------------------------------------------------------------
__global__ void mha_cvt_f16(const float* __restrict__ src,
                            _Float16* __restrict__ dst, int n) {
    int i = blockIdx.x * blockDim.x + threadIdx.x;
    if (i < n) dst[i] = (_Float16)src[i];
}

// ---------------------------------------------------------------------------
// Kernel 2: QKV projection. C[s,e] = sum_d q[s,d]*Win[e,d] + bias[e].
// 32x64 tile per wave: 2 A-frags x 4 B-frags -> 8 WMMA per 32-deep k-step.
// Epilogue scatters into Qh[h][s][hd], Kh[h][s][hd], Vt[h][hd][s] (f16);
// each wave's 64 columns lie in exactly one of Q/K/V and one head.
// ---------------------------------------------------------------------------
__global__ void mha_qkv_gemm(const _Float16* __restrict__ qf,
                             const _Float16* __restrict__ wIn,
                             const float* __restrict__ bias,
                             _Float16* __restrict__ Qh,
                             _Float16* __restrict__ Kh,
                             _Float16* __restrict__ Vt) {
    const int lane = threadIdx.x & 31;
    const int wave = threadIdx.x >> 5;
    const int gw   = blockIdx.x * 8 + wave;
    const int ti   = gw / (E3 / 64);      // row tile (32 rows)
    const int tj   = gw % (E3 / 64);      // col tile (64 cols)
    const int s0   = ti * 32;
    const int e0   = tj * 64;
    const int n    = lane & 15;
    const int hi   = lane >> 4;
    const int klo  = hi * 8;
    const int kb   = hi * 16;

    const _Float16* arow0 = qf + (size_t)(s0 + n) * D_DIM;       // rows 0..15
    const _Float16* arow1 = qf + (size_t)(s0 + 16 + n) * D_DIM;  // rows 16..31
    const _Float16* brow  = wIn + (size_t)(e0 + n) * D_DIM + kb; // col n of j=0

    f8 acc[2][4];
    #pragma unroll
    for (int i = 0; i < 2; ++i)
        #pragma unroll
        for (int j = 0; j < 4; ++j) acc[i][j] = (f8){};

    #pragma unroll 2
    for (int d0 = 0; d0 < D_DIM; d0 += 32) {
        h16 a0 = load_a(arow0 + d0, klo);
        h16 a1 = load_a(arow1 + d0, klo);
        #pragma unroll
        for (int j = 0; j < 4; ++j) {
            h16 b = *(const h16*)(brow + (size_t)(16 * j) * D_DIM + d0);
            acc[0][j] = wmma_f16(a0, b, acc[0][j]);
            acc[1][j] = wmma_f16(a1, b, acc[1][j]);
        }
    }

    // 64-col span is 64-aligned -> single c (Q/K/V) and single head.
    const int c = e0 >> 9;
    const int h = (e0 & 511) >> 6;
    float bv[4];
    #pragma unroll
    for (int j = 0; j < 4; ++j) bv[j] = bias[e0 + 16 * j + n];

    if (c == 2) {
        // Vt[(h*HD + hd)*S + s]; s = s0 + 16*i + 8*hi + r contiguous over r.
        #pragma unroll
        for (int i = 0; i < 2; ++i)
            #pragma unroll
            for (int j = 0; j < 4; ++j) {
                h8 v;
                #pragma unroll
                for (int r = 0; r < 8; ++r) v[r] = (_Float16)(acc[i][j][r] + bv[j]);
                *(h8*)(Vt + (size_t)(h * HD + 16 * j + n) * S_LEN
                          + s0 + 16 * i + 8 * hi) = v;
            }
    } else {
        _Float16* base = (c == 0) ? Qh : Kh;
        #pragma unroll
        for (int i = 0; i < 2; ++i)
            #pragma unroll
            for (int j = 0; j < 4; ++j)
                #pragma unroll
                for (int r = 0; r < 8; ++r) {
                    int s = s0 + 16 * i + 8 * hi + r;
                    base[((size_t)h * S_LEN + s) * HD + 16 * j + n] =
                        (_Float16)(acc[i][j][r] + bv[j]);
                }
    }
}

// ---------------------------------------------------------------------------
// Kernel 3: flash attention. One wave = 16 query rows; all 8 waves of a
// block share one head, so K (32x64) and Vt (64x32) tiles are staged in LDS
// once per block per 32-key step, double-buffered (one barrier per step).
// Per step and wave: 4 WMMA (scores) + online softmax + 4 WMMA (P*V).
// P re-layout (C-layout f32 -> A-layout f16) goes through a per-wave LDS
// region; LDS ops from one wave are in-order, so only a scheduling fence.
// ---------------------------------------------------------------------------
__global__ void mha_flash_attn(const _Float16* __restrict__ Qh,
                               const _Float16* __restrict__ Kh,
                               const _Float16* __restrict__ Vt,
                               _Float16* __restrict__ O) {
    __shared__ alignas(16) _Float16 Ks[2][32][64];   // [buf][key][d]   8 KB
    __shared__ alignas(16) _Float16 Vs[2][64][32];   // [buf][hd][key]  8 KB
    __shared__ alignas(16) _Float16 Pl[8][16][32];   // per-wave P      8 KB

    const int tid  = threadIdx.x;
    const int lane = tid & 31;
    const int wave = tid >> 5;
    const int gw   = blockIdx.x * 8 + wave;
    const int h    = gw >> 8;             // head (uniform per block)
    const int qt   = gw & 255;
    const int s0   = qt * 16;
    const int n    = lane & 15;
    const int hi   = lane >> 4;
    const int klo  = hi * 8;
    const int kb   = hi * 16;
    const float scale = 0.125f;           // 1/sqrt(64)

    const _Float16* Kb = Kh + (size_t)h * S_LEN * HD;
    const _Float16* Vb = Vt + (size_t)h * HD * S_LEN;

    // Q fragments (Hd=64 -> two 16x32 A-frags) held in registers throughout.
    const _Float16* qrow = Qh + ((size_t)h * S_LEN + s0 + n) * HD;
    const h16 aq0 = load_a(qrow,      klo);
    const h16 aq1 = load_a(qrow + 32, klo);

    // Cooperative stage of one 32-key step into LDS buffer `buf`.
    // K tile is contiguous in global (32 rows x 64 halves); V tile is 64
    // rows of 32 contiguous halves at stride S. 16 B per thread per tile.
    auto stage = [&](int buf, int j) {
        stage16(Kb + (size_t)j * HD + 8 * tid, &Ks[buf][0][0] + 8 * tid);
        int vr = tid >> 2, vc = (tid & 3) * 8;
        stage16(Vb + (size_t)vr * S_LEN + j + vc, &Vs[buf][vr][vc]);
    };

    float mrow[8], lrow[8];
    f8 acc[4];
    #pragma unroll
    for (int r = 0; r < 8; ++r) { mrow[r] = -3.0e38f; lrow[r] = 0.f; }
    #pragma unroll
    for (int t = 0; t < 4; ++t) acc[t] = (f8){};

    stage(0, 0);
    for (int it = 0; it < S_LEN / 32; ++it) {
        const int cur = it & 1;
        stage_wait();                          // my async stores to LDS done
        __syncthreads();                       // cur staged; prev reads done
        if (it + 1 < S_LEN / 32) stage(cur ^ 1, (it + 1) * 32);

        // ---- scores: 16 queries x 32 keys, contraction over Hd=64 ----
        f8 sc0 = {}, sc1 = {};
        sc0 = wmma_f16(aq0, *(const h16*)(&Ks[cur][n][kb]),           sc0);
        sc0 = wmma_f16(aq1, *(const h16*)(&Ks[cur][n][32 + kb]),      sc0);
        sc1 = wmma_f16(aq0, *(const h16*)(&Ks[cur][16 + n][kb]),      sc1);
        sc1 = wmma_f16(aq1, *(const h16*)(&Ks[cur][16 + n][32 + kb]), sc1);

        // ---- online softmax over the 32 new keys ----
        float cf[8];
        #pragma unroll
        for (int r = 0; r < 8; ++r) {
            float v0 = sc0[r] * scale;
            float v1 = sc1[r] * scale;
            float mx = fmaxf(v0, v1);
            mx = fmaxf(mx, __shfl_xor(mx, 1));
            mx = fmaxf(mx, __shfl_xor(mx, 2));
            mx = fmaxf(mx, __shfl_xor(mx, 4));
            mx = fmaxf(mx, __shfl_xor(mx, 8));     // row max (16-lane group)
            float mnew = fmaxf(mrow[r], mx);
            cf[r] = __expf(mrow[r] - mnew);
            mrow[r] = mnew;
            float p0 = __expf(v0 - mnew);
            float p1 = __expf(v1 - mnew);
            sc0[r] = p0; sc1[r] = p1;
            float rs = p0 + p1;
            rs += __shfl_xor(rs, 1);
            rs += __shfl_xor(rs, 2);
            rs += __shfl_xor(rs, 4);
            rs += __shfl_xor(rs, 8);               // row sum
            lrow[r] = lrow[r] * cf[r] + rs;
        }
        #pragma unroll
        for (int t = 0; t < 4; ++t)
            #pragma unroll
            for (int r = 0; r < 8; ++r) acc[t][r] *= cf[r];

        // ---- re-layout P (C-layout f32) -> A-layout f16 via LDS ----
        #pragma unroll
        for (int r = 0; r < 8; ++r) {
            int m = r + 8 * hi;
            Pl[wave][m][n]      = (_Float16)sc0[r];
            Pl[wave][m][n + 16] = (_Float16)sc1[r];
        }
        __builtin_amdgcn_wave_barrier();
        h16 ap = cat16(*(const h8*)&Pl[wave][n][kb],
                       *(const h8*)&Pl[wave][n][kb + 8]);
        __builtin_amdgcn_wave_barrier();

        // ---- out += P(16x32) x V(32x64): 4 WMMA over Hd tiles ----
        #pragma unroll
        for (int t = 0; t < 4; ++t)
            acc[t] = wmma_f16(ap, *(const h16*)(&Vs[cur][t * 16 + n][kb]), acc[t]);
    }

    // ---- normalize + store O[s][h*64 + hd] as f16 ----
    #pragma unroll
    for (int t = 0; t < 4; ++t)
        #pragma unroll
        for (int r = 0; r < 8; ++r) {
            int m = r + 8 * hi;
            float o = acc[t][r] / lrow[r];
            O[(size_t)(s0 + m) * D_DIM + h * HD + t * 16 + n] = (_Float16)o;
        }
}

// ---------------------------------------------------------------------------
// Kernel 4: output projection. out[s,e] = sum_d O[s,d]*Wo[e,d] + bias[e].
// Same 32x64-per-wave tiling; f32 output straight to d_out.
// ---------------------------------------------------------------------------
__global__ void mha_out_gemm(const _Float16* __restrict__ Of,
                             const _Float16* __restrict__ wOut,
                             const float* __restrict__ bias,
                             float* __restrict__ out) {
    const int lane = threadIdx.x & 31;
    const int wave = threadIdx.x >> 5;
    const int gw   = blockIdx.x * 8 + wave;
    const int ti   = gw / (D_DIM / 64);
    const int tj   = gw % (D_DIM / 64);
    const int s0   = ti * 32;
    const int e0   = tj * 64;
    const int n    = lane & 15;
    const int hi   = lane >> 4;
    const int klo  = hi * 8;
    const int kb   = hi * 16;

    const _Float16* arow0 = Of + (size_t)(s0 + n) * D_DIM;
    const _Float16* arow1 = Of + (size_t)(s0 + 16 + n) * D_DIM;
    const _Float16* brow  = wOut + (size_t)(e0 + n) * D_DIM + kb;

    f8 acc[2][4];
    #pragma unroll
    for (int i = 0; i < 2; ++i)
        #pragma unroll
        for (int j = 0; j < 4; ++j) acc[i][j] = (f8){};

    #pragma unroll 2
    for (int d0 = 0; d0 < D_DIM; d0 += 32) {
        h16 a0 = load_a(arow0 + d0, klo);
        h16 a1 = load_a(arow1 + d0, klo);
        #pragma unroll
        for (int j = 0; j < 4; ++j) {
            h16 b = *(const h16*)(brow + (size_t)(16 * j) * D_DIM + d0);
            acc[0][j] = wmma_f16(a0, b, acc[0][j]);
            acc[1][j] = wmma_f16(a1, b, acc[1][j]);
        }
    }

    #pragma unroll
    for (int j = 0; j < 4; ++j) {
        float bv = bias[e0 + 16 * j + n];
        #pragma unroll
        for (int i = 0; i < 2; ++i)
            #pragma unroll
            for (int r = 0; r < 8; ++r) {
                int s = s0 + 16 * i + 8 * hi + r;
                out[(size_t)s * D_DIM + e0 + 16 * j + n] = acc[i][j][r] + bv;
            }
    }
}

// ---------------------------------------------------------------------------
// Launch
// ---------------------------------------------------------------------------
extern "C" void kernel_launch(void* const* d_in, const int* in_sizes, int n_in,
                              void* d_out, int out_size, void* d_ws, size_t ws_size,
                              hipStream_t stream) {
    (void)in_sizes; (void)n_in; (void)out_size; (void)ws_size;

    const float* query    = (const float*)d_in[0]; // [S, D]
    const float* wIn_f32  = (const float*)d_in[1]; // [3D, D]
    const float* bIn      = (const float*)d_in[2]; // [3D]
    const float* wOut_f32 = (const float*)d_in[3]; // [D, D]
    const float* bOut     = (const float*)d_in[4]; // [D]
    float* out            = (float*)d_out;         // [S, D]

    char* ws = (char*)d_ws;
    size_t off = 0;
    _Float16* qf   = (_Float16*)(ws + off); off += (size_t)S_LEN * D_DIM * 2;  // 4 MiB
    _Float16* wIn  = (_Float16*)(ws + off); off += (size_t)E3 * D_DIM * 2;     // 1.5 MiB
    _Float16* wOut = (_Float16*)(ws + off); off += (size_t)D_DIM * D_DIM * 2;  // 0.5 MiB
    _Float16* Qh   = (_Float16*)(ws + off); off += (size_t)H_NUM * S_LEN * HD * 2;
    _Float16* Kh   = (_Float16*)(ws + off); off += (size_t)H_NUM * S_LEN * HD * 2;
    _Float16* Vt   = (_Float16*)(ws + off); off += (size_t)H_NUM * HD * S_LEN * 2;
    _Float16* Of   = (_Float16*)(ws + off); off += (size_t)S_LEN * D_DIM * 2;

    const int nq  = S_LEN * D_DIM;       // 2097152
    const int nwi = E3 * D_DIM;          // 786432
    const int nwo = D_DIM * D_DIM;       // 262144
    mha_cvt_f16<<<(nq  + 255) / 256, 256, 0, stream>>>(query,    qf,   nq);
    mha_cvt_f16<<<(nwi + 255) / 256, 256, 0, stream>>>(wIn_f32,  wIn,  nwi);
    mha_cvt_f16<<<(nwo + 255) / 256, 256, 0, stream>>>(wOut_f32, wOut, nwo);

    // QKV GEMM: (S/32)*(3D/64) = 128*24 = 3072 waves, 8 waves/block.
    mha_qkv_gemm<<<3072 / 8, 256, 0, stream>>>(qf, wIn, bIn, Qh, Kh, Vt);

    // Flash attention: H * (S/16) = 2048 waves, 8 waves/block (one head/block).
    mha_flash_attn<<<2048 / 8, 256, 0, stream>>>(Qh, Kh, Vt, Of);

    // Out projection: (S/32)*(D/64) = 128*8 = 1024 waves, 8 waves/block.
    mha_out_gemm<<<1024 / 8, 256, 0, stream>>>(Of, wOut, bOut, out);
}
